// TransformerBlock_24352464569483
// MI455X (gfx1250) — matrix operand order repaired and checked by reading, not compiled, not run on hardware
//
#include <hip/hip_runtime.h>

typedef __attribute__((ext_vector_type(16))) __bf16 v16bf;
typedef __attribute__((ext_vector_type(8)))  float  v8f;

// ---------------------------------------------------------------------------
// helpers
// ---------------------------------------------------------------------------
__device__ __forceinline__ v8f vzero() {
  v8f z;
#pragma unroll
  for (int r = 0; r < 8; ++r) z[r] = 0.f;
  return z;
}

__device__ __forceinline__ v8f wmma_bf16(v16bf a, v16bf b, v8f c) {
  // D = A(16x32 bf16) * B(32x16 bf16) + C(16x16 f32)
  return __builtin_amdgcn_wmma_f32_16x16x32_bf16(false, a, false, b, (short)0, c,
                                                 false, false);
}

// A-operand K index for element i of v16bf (contiguous: k = 16*hf + i, matches B)
__device__ __forceinline__ int a_kidx(int i, int hf) {
  return i + 16 * hf;
}

// ---- async global -> LDS copy (CDNA5 GLOBAL_LOAD_ASYNC_TO_LDS_B32) --------
#if __has_builtin(__builtin_amdgcn_global_load_async_to_lds_b32)
__device__ __forceinline__ void async_cp_b32(const float* g, float* l) {
  __builtin_amdgcn_global_load_async_to_lds_b32(
      (__attribute__((address_space(1))) int*)g,
      (__attribute__((address_space(3))) int*)l, 0, 0);
}
__device__ __forceinline__ void async_wait_all() {
#if __has_builtin(__builtin_amdgcn_s_wait_asynccnt)
  __builtin_amdgcn_s_wait_asynccnt(0);
#else
  asm volatile("s_wait_asynccnt 0x0" ::: "memory");
#endif
}
#else
__device__ __forceinline__ void async_cp_b32(const float* g, float* l) { *l = *g; }
__device__ __forceinline__ void async_wait_all() {}
#endif

// ---------------------------------------------------------------------------
// BatchNorm: per-channel mean / inv-std over (N,H,W)
// ---------------------------------------------------------------------------
__global__ void bn_stats_kernel(const float* __restrict__ x, float* __restrict__ stats,
                                int Cc, int HWc, int Bn) {
  const int c = blockIdx.x;
  __shared__ float ss[256];
  __shared__ float sq[256];
  float s = 0.f, q = 0.f;
  const int total = Bn * HWc;
  for (int i = threadIdx.x; i < total; i += blockDim.x) {
    int b = i / HWc, o = i - b * HWc;
    float v = x[(b * Cc + c) * HWc + o];
    s += v;
    q += v * v;
  }
  ss[threadIdx.x] = s;
  sq[threadIdx.x] = q;
  __syncthreads();
  for (int st = 128; st > 0; st >>= 1) {
    if ((int)threadIdx.x < st) {
      ss[threadIdx.x] += ss[threadIdx.x + st];
      sq[threadIdx.x] += sq[threadIdx.x + st];
    }
    __syncthreads();
  }
  if (threadIdx.x == 0) {
    float mean = ss[0] / (float)total;
    float var  = sq[0] / (float)total - mean * mean;
    stats[2 * c]     = mean;
    stats[2 * c + 1] = rsqrtf(var + 1e-5f);
  }
}

// act: 0 none, 1 leaky(0.2), 2 relu ; optional residual add after activation
__global__ void bn_apply_kernel(const float* __restrict__ x, const float* __restrict__ stats,
                                const float* __restrict__ g, const float* __restrict__ bt,
                                const float* __restrict__ res, float* __restrict__ y,
                                int Cc, int HWc, int total, int act) {
  int i = blockIdx.x * blockDim.x + threadIdx.x;
  if (i >= total) return;
  int c = (i / HWc) % Cc;
  float v = g[c] * (x[i] - stats[2 * c]) * stats[2 * c + 1] + bt[c];
  if (act == 1) v = (v >= 0.f) ? v : 0.2f * v;
  else if (act == 2) v = fmaxf(v, 0.f);
  y[i] = (res ? res[i] : 0.f) + v;
}

__global__ void add_kernel(const float* __restrict__ a, const float* __restrict__ b,
                           float* __restrict__ y, int n) {
  int i = blockIdx.x * blockDim.x + threadIdx.x;
  if (i < n) y[i] = a[i] + b[i];
}

// ---------------------------------------------------------------------------
// Implicit-GEMM convolution on WMMA with LDS staging via async copies.
// Block = 128 threads (4 waves) computing a 64(co) x 64(position) macro-tile.
// Per (kh,kw,ci-chunk of 32): stage X[32ci x 64pos] and W[64co x 32ci] in LDS
// (async global->LDS, ds_store zero-fill for padding), then each wave does
// 4 WMMAs on its 16-co sub-tile. Optional channel concat via X2/csplit.
// ---------------------------------------------------------------------------
__global__ void __launch_bounds__(128)
conv_wmma_kernel(const float* __restrict__ X, const float* __restrict__ X2, int csplit,
                 const float* __restrict__ Wt, const float* __restrict__ bias,
                 float* __restrict__ Y,
                 int Cin, int Cout, int Hin, int Win, int Hout, int Wout,
                 int KH, int KW, int stride, int pad, int dil) {
  __shared__ float sX[32 * 64];  // [ci][pos]
  __shared__ float sW[64 * 32];  // [co][ci]

  const int t = threadIdx.x;
  const int lane = t & 31, wave = t >> 5;
  const int m = lane & 15, hf = lane >> 4;
  const int b = blockIdx.z;
  const int co0 = blockIdx.y * 64;
  const int posBase = blockIdx.x * 64;
  const int HWo = Hout * Wout;

  // per-thread cooperative-load coordinates (fixed across chunks)
  const int xpos = t & 63;   // position column in sX
  const int xciB = t >> 6;   // 0/1: base ci parity
  const int n_x = posBase + xpos;
  const bool nok_x = n_x < HWo;
  const int oh_x = nok_x ? n_x / Wout : 0;
  const int ow_x = nok_x ? n_x - (n_x / Wout) * Wout : 0;
  const int wci = t & 31;    // ci column in sW
  const int wcoB = t >> 5;   // 0..3: base co

  v8f acc[4];
#pragma unroll
  for (int p = 0; p < 4; ++p) acc[p] = vzero();

  for (int kh = 0; kh < KH; ++kh) {
    const int ih = oh_x * stride - pad + kh * dil;
    for (int kw = 0; kw < KW; ++kw) {
      const int iw = ow_x * stride - pad + kw * dil;
      const bool inb =
          nok_x && (unsigned)ih < (unsigned)Hin && (unsigned)iw < (unsigned)Win;
      for (int c0 = 0; c0 < Cin; c0 += 32) {
        // ---- stage activation tile ----
#pragma unroll
        for (int j = 0; j < 16; ++j) {
          int cil = xciB + 2 * j;  // 0..31
          float* dst = &sX[cil * 64 + xpos];
          if (inb) {
            int ci = c0 + cil;
            const float* src = X;
            int cc = ci, cw = X2 ? csplit : Cin;
            if (X2 && ci >= csplit) { src = X2; cc = ci - csplit; cw = Cin - csplit; }
            async_cp_b32(src + ((b * cw + cc) * Hin + ih) * Win + iw, dst);
          } else {
            *dst = 0.f;
          }
        }
        // ---- stage weight tile ----
#pragma unroll
        for (int j = 0; j < 16; ++j) {
          int col = wcoB + 4 * j;  // 0..63
          async_cp_b32(Wt + (((co0 + col) * Cin + c0 + wci) * KH + kh) * KW + kw,
                       &sW[col * 32 + wci]);
        }
        async_wait_all();
        __syncthreads();

        // ---- compute: wave owns co sub-tile [wave*16, wave*16+16) ----
        v16bf af;
#pragma unroll
        for (int i = 0; i < 16; ++i)
          af[i] = (__bf16)sW[(wave * 16 + m) * 32 + a_kidx(i, hf)];
#pragma unroll
        for (int p = 0; p < 4; ++p) {
          v16bf bx;
#pragma unroll
          for (int i = 0; i < 16; ++i)
            bx[i] = (__bf16)sX[(i + 16 * hf) * 64 + p * 16 + m];
          acc[p] = wmma_bf16(af, bx, acc[p]);
        }
        __syncthreads();
      }
    }
  }
#pragma unroll
  for (int p = 0; p < 4; ++p) {
#pragma unroll
    for (int r = 0; r < 8; ++r) {
      int co = co0 + wave * 16 + r + 8 * hf;
      int n = posBase + p * 16 + m;
      if (n < HWo) Y[(b * Cout + co) * HWo + n] = acc[p][r] + (bias ? bias[co] : 0.f);
    }
  }
}

// ---------------------------------------------------------------------------
// Windowed multi-scale attention (one wave per 16-query tile of one window,
// dk = 32, flash-style online softmax; P relaid through LDS into A layout).
// Writes Y[ch oBase..oBase+31] = XC + attention output.
// ---------------------------------------------------------------------------
__global__ void __launch_bounds__(32)
grid_attn_kernel(const float* __restrict__ Q, int qC, int qBase,
                 const float* __restrict__ K, int kC, int kBase,
                 const float* __restrict__ V, int vC, int vBase,
                 const float* __restrict__ XC, float* __restrict__ Y,
                 int Cc, int oBase, int Hh, int Ww, int ph, int pw,
                 int ntok, int qtiles, float scale) {
  const int lane = threadIdx.x;
  const int m  = lane & 15;
  const int hf = lane >> 4;
  const int b   = blockIdx.y;
  const int win = blockIdx.x / qtiles;
  const int qt  = blockIdx.x - win * qtiles;
  const int gw = Ww / pw;
  const int gy = win / gw, gx = win - gy * gw;
  const int HWl = Hh * Ww;

  // Q tile (A operand), pre-scaled
  const int qrow = qt * 16 + m;
  const bool qok = qrow < ntok;
  const int qh = gy * ph + (qok ? qrow / pw : 0);
  const int qw = gx * pw + (qok ? qrow - (qrow / pw) * pw : 0);
  v16bf aq;
#pragma unroll
  for (int i = 0; i < 16; ++i) {
    int d = a_kidx(i, hf);
    float v = qok ? Q[((b * qC + qBase + d) * Hh + qh) * Ww + qw] * scale : 0.f;
    aq[i] = (__bf16)v;
  }

  v8f o0 = vzero(), o1 = vzero();
  float mrow = -1e30f, lrow = 0.f;  // stats for row = lane&15
  const int rr = m & 7, shl = (m >> 3) << 4;
  __shared__ __bf16 sP[16 * 32];

  for (int k0 = 0; k0 < ntok; k0 += 32) {
    v8f s0, s1;
    // S tiles: keys [k0,k0+16), [k0+16,k0+32)
#pragma unroll
    for (int t = 0; t < 2; ++t) {
      v16bf bk;
      int key = k0 + t * 16 + m;
      bool ok = key < ntok;
      int kh = gy * ph + (ok ? key / pw : 0);
      int kw = gx * pw + (ok ? key - (key / pw) * pw : 0);
#pragma unroll
      for (int i = 0; i < 16; ++i) {
        int d = i + 16 * hf;
        float v = ok ? K[((b * kC + kBase + d) * Hh + kh) * Ww + kw] : 0.f;
        bk[i] = (__bf16)v;
      }
      if (t == 0) s0 = wmma_bf16(aq, bk, vzero());
      else        s1 = wmma_bf16(aq, bk, vzero());
    }
    // mask invalid key columns
    bool kinv0 = (k0 + m) >= ntok;
    bool kinv1 = (k0 + 16 + m) >= ntok;
#pragma unroll
    for (int r = 0; r < 8; ++r) {
      if (kinv0) s0[r] = -1e30f;
      if (kinv1) s1[r] = -1e30f;
    }
    // tile row max -> per-lane stat for row lane&15
    float tm = -1e30f;
#pragma unroll
    for (int r = 0; r < 8; ++r) {
      float v = fmaxf(s0[r], s1[r]);
#pragma unroll
      for (int off = 1; off < 16; off <<= 1) v = fmaxf(v, __shfl_xor(v, off, 32));
      float g = __shfl(v, shl, 32);
      if (r == rr) tm = g;
    }
    float mnew = fmaxf(mrow, tm);
    float corr = __expf(mrow - mnew);
    // P = exp(S - m), tile row sums
    float ts = 0.f;
#pragma unroll
    for (int r = 0; r < 8; ++r) {
      float mr = __shfl(mnew, r + 8 * hf, 32);
      float p0 = __expf(s0[r] - mr);
      float p1 = __expf(s1[r] - mr);
      s0[r] = p0; s1[r] = p1;
      float v = p0 + p1;
#pragma unroll
      for (int off = 1; off < 16; off <<= 1) v += __shfl_xor(v, off, 32);
      float g = __shfl(v, shl, 32);
      if (r == rr) ts = g;
    }
    lrow = lrow * corr + ts;
    mrow = mnew;
    // rescale O accumulators by row correction
#pragma unroll
    for (int r = 0; r < 8; ++r) {
      float cr = __shfl(corr, r + 8 * hf, 32);
      o0[r] *= cr; o1[r] *= cr;
    }
    // relay P (C/D layout) through LDS into A layout
    __syncthreads();
#pragma unroll
    for (int r = 0; r < 8; ++r) {
      sP[(r + 8 * hf) * 32 + m]      = (__bf16)s0[r];
      sP[(r + 8 * hf) * 32 + 16 + m] = (__bf16)s1[r];
    }
    __syncthreads();
    v16bf ap;
#pragma unroll
    for (int i = 0; i < 16; ++i) ap[i] = sP[m * 32 + a_kidx(i, hf)];
    // V tiles: B operand, N = channel d (two 16-wide halves)
    v16bf bv0, bv1;
#pragma unroll
    for (int i = 0; i < 16; ++i) {
      int key = k0 + i + 16 * hf;
      float v0 = 0.f, v1 = 0.f;
      if (key < ntok) {
        int kh = gy * ph + key / pw;
        int kw = gx * pw + (key - (key / pw) * pw);
        int base = ((b * vC + vBase) * Hh + kh) * Ww + kw;
        v0 = V[base + m * HWl];
        v1 = V[base + (16 + m) * HWl];
      }
      bv0[i] = (__bf16)v0;
      bv1[i] = (__bf16)v1;
    }
    o0 = wmma_bf16(ap, bv0, o0);
    o1 = wmma_bf16(ap, bv1, o1);
  }

  // normalize + residual + store
#pragma unroll
  for (int r = 0; r < 8; ++r) {
    int row = r + 8 * hf;
    float li = __shfl(lrow, row, 32);
    float inv = 1.f / li;
    int q = qt * 16 + row;
    if (q < ntok) {
      int oh = gy * ph + q / pw;
      int ow = gx * pw + (q - (q / pw) * pw);
      int base = ((b * Cc + oBase) * Hh + oh) * Ww + ow;
      Y[base + m * HWl]        = XC[base + m * HWl] + o0[r] * inv;
      Y[base + (16 + m) * HWl] = XC[base + (16 + m) * HWl] + o1[r] * inv;
    }
  }
}

// ---------------------------------------------------------------------------
// Global spatial-reduction attention: heads=16, hd=8, 64 keys (K-dim padded
// to 32 with zeros). One wave per (b, head, 16-query tile).
// ---------------------------------------------------------------------------
__global__ void __launch_bounds__(32)
global_attn_kernel(const float* __restrict__ Qt, const float* __restrict__ KVt,
                   float* __restrict__ Y, int Cc, int HWq, int L, int hd, float scale) {
  const int lane = threadIdx.x;
  const int m  = lane & 15;
  const int hf = lane >> 4;
  const int b = blockIdx.z, head = blockIdx.y, qt = blockIdx.x;
  const int q = qt * 16 + m;

  v16bf aq;
#pragma unroll
  for (int i = 0; i < 16; ++i) {
    int d = a_kidx(i, hf);
    float v = (d < hd) ? Qt[(b * Cc + head * hd + d) * HWq + q] * scale : 0.f;
    aq[i] = (__bf16)v;
  }
  v8f s[4];
#pragma unroll
  for (int kt = 0; kt < 4; ++kt) {
    v16bf bk;
#pragma unroll
    for (int i = 0; i < 16; ++i) {
      int d = i + 16 * hf;
      float v = (d < hd) ? KVt[(b * 2 * Cc + head * hd + d) * L + kt * 16 + m] : 0.f;
      bk[i] = (__bf16)v;
    }
    s[kt] = wmma_bf16(aq, bk, vzero());
  }
  const int rr = m & 7, shl = (m >> 3) << 4;
  float mrow = -1e30f;
#pragma unroll
  for (int r = 0; r < 8; ++r) {
    float v = fmaxf(fmaxf(s[0][r], s[1][r]), fmaxf(s[2][r], s[3][r]));
#pragma unroll
    for (int off = 1; off < 16; off <<= 1) v = fmaxf(v, __shfl_xor(v, off, 32));
    float g = __shfl(v, shl, 32);
    if (r == rr) mrow = g;
  }
  float lrow = 0.f;
#pragma unroll
  for (int r = 0; r < 8; ++r) {
    float mr = __shfl(mrow, r + 8 * hf, 32);
    float v = 0.f;
#pragma unroll
    for (int kt = 0; kt < 4; ++kt) {
      float p = __expf(s[kt][r] - mr);
      s[kt][r] = p;
      v += p;
    }
#pragma unroll
    for (int off = 1; off < 16; off <<= 1) v += __shfl_xor(v, off, 32);
    float g = __shfl(v, shl, 32);
    if (r == rr) lrow = g;
  }
  __shared__ __bf16 sP[16 * 64];
#pragma unroll
  for (int r = 0; r < 8; ++r) {
    int row = r + 8 * hf;
#pragma unroll
    for (int kt = 0; kt < 4; ++kt) sP[row * 64 + kt * 16 + m] = (__bf16)s[kt][r];
  }
  __syncthreads();
  v8f o = vzero();
#pragma unroll
  for (int c2 = 0; c2 < 2; ++c2) {
    v16bf ap, bv;
#pragma unroll
    for (int i = 0; i < 16; ++i) ap[i] = sP[m * 64 + c2 * 32 + a_kidx(i, hf)];
#pragma unroll
    for (int i = 0; i < 16; ++i) {
      int key = c2 * 32 + i + 16 * hf;
      float v = (m < hd) ? KVt[(b * 2 * Cc + Cc + head * hd + m) * L + key] : 0.f;
      bv[i] = (__bf16)v;
    }
    o = wmma_bf16(ap, bv, o);
  }
#pragma unroll
  for (int r = 0; r < 8; ++r) {
    int row = r + 8 * hf;
    float li = __shfl(lrow, row, 32);
    int qrow = qt * 16 + row;
    if (m < hd) Y[(b * Cc + head * hd + m) * HWq + qrow] = o[r] / li;
  }
}

// ---------------------------------------------------------------------------
// Host orchestration
// ---------------------------------------------------------------------------
extern "C" void kernel_launch(void* const* d_in, const int* in_sizes, int n_in,
                              void* d_out, int out_size, void* d_ws, size_t ws_size,
                              hipStream_t stream) {
  enum {
    I_RGB = 0, I_FRE,
    LA1_NG, LA1_NB, LA1_QV_W, LA1_QV_B, LA1_K_W, LA1_K_B, LA1_OUT_W, LA1_OUT_G, LA1_OUT_BT,
    LA2_NG, LA2_NB, LA2_QK_W, LA2_QK_B, LA2_V_W, LA2_V_B, LA2_OUT_W, LA2_OUT_G, LA2_OUT_BT,
    GA_Q_W, GA_Q_B, GA_SR_W, GA_SR_B, GA_NG, GA_NB, GA_KV_W, GA_KV_B, GA_OUT_W, GA_OUT_G, GA_OUT_BT,
    CV_W1, CV_G1, CV_BT1, CV_W2, CV_G2, CV_BT2,
    FF1_W1, FF1_B1, FF1_G1, FF1_BT1, FF1_W2, FF1_B2, FF1_G2, FF1_BT2,
    FF2_W1, FF2_B1, FF2_G1, FF2_BT1, FF2_W2, FF2_B2, FF2_G2, FF2_BT2
  };
  const float* P[64];
  for (int i = 0; i < n_in && i < 64; ++i) P[i] = (const float*)d_in[i];
  const float* rgb = P[I_RGB];
  const float* fre = P[I_FRE];

  const int Bn = 16, C = 128, Hh = 56, Ww = 56, HW = Hh * Ww;
  const int T = Bn * C * HW;
  const float SCALE32 = 0.17677669529663687f;  // 32^-0.5
  const float SCALE8  = 0.35355339059327373f;  // 8^-0.5

  float* ws   = (float*)d_ws;
  float* A_   = ws;            // T
  float* QV   = A_ + T;        // 2T
  float* K1   = QV + 2 * T;    // T
  float* ATT1 = K1 + T;        // T
  float* ATT2 = ATT1 + T;      // T
  float* OUTB = ATT2 + T;      // T
  float* TMP  = OUTB + T;      // T
  float* TMP2 = TMP + T;       // T
  float* SR   = TMP2 + T;      // 16*128*64
  float* KV   = SR + Bn * C * 64;      // 16*256*64
  float* ST   = KV + Bn * 2 * C * 64;  // 256

  auto bn = [&](const float* x, const float* g, const float* bt, const float* res,
                float* y, int Cc, int HWc, int act) {
    bn_stats_kernel<<<Cc, 256, 0, stream>>>(x, ST, Cc, HWc, Bn);
    int tot = Bn * Cc * HWc;
    bn_apply_kernel<<<(tot + 255) / 256, 256, 0, stream>>>(x, ST, g, bt, res, y, Cc, HWc, tot, act);
  };
  auto conv = [&](const float* X, const float* X2, int csplit, const float* Wt,
                  const float* bias, float* Y, int Cin, int Cout, int Hin, int Win,
                  int Hout, int Wout, int KH, int KW, int stride, int pad, int dil) {
    dim3 grid((Hout * Wout + 63) / 64, Cout / 64, Bn);
    conv_wmma_kernel<<<grid, 128, 0, stream>>>(X, X2, csplit, Wt, bias, Y, Cin, Cout,
                                               Hin, Win, Hout, Wout, KH, KW, stride, pad, dil);
  };
  auto multi_attn = [&](const float* Q, int qC, int qOff, const float* K, int kC, int kOff,
                        const float* V, int vC, int vOff, float* Y) {
    const int PW[4] = {28, 14, 8, 7};
    const int PH[4] = {28, 14, 8, 7};
    for (int g4 = 0; g4 < 4; ++g4) {
      int ph = PH[g4], pw = PW[g4];
      int gh = Hh / ph, gw = Ww / pw;
      int ntok = ph * pw;
      int qt = (ntok + 15) / 16;
      dim3 grid(gh * gw * qt, Bn);
      grid_attn_kernel<<<grid, 32, 0, stream>>>(
          Q, qC, qOff + g4 * 32, K, kC, kOff + g4 * 32, V, vC, vOff + g4 * 32,
          rgb, Y, C, g4 * 32, Hh, Ww, ph, pw, ntok, qt, SCALE32);
    }
  };
  auto addv = [&](const float* a, const float* b, float* y) {
    add_kernel<<<(T + 255) / 256, 256, 0, stream>>>(a, b, y, T);
  };

  // ---- local attention 1 (q,v from rgb ; k from fre) ----
  bn(rgb, P[LA1_NG], P[LA1_NB], nullptr, A_, C, HW, 0);
  conv(A_, nullptr, 0, P[LA1_QV_W], P[LA1_QV_B], QV, C, 2 * C, Hh, Ww, Hh, Ww, 1, 1, 1, 0, 1);
  bn(fre, P[LA1_NG], P[LA1_NB], nullptr, A_, C, HW, 0);
  conv(A_, nullptr, 0, P[LA1_K_W], P[LA1_K_B], K1, C, C, Hh, Ww, Hh, Ww, 1, 1, 1, 0, 1);
  multi_attn(QV, 2 * C, 0, K1, C, 0, QV, 2 * C, C, TMP);
  conv(TMP, nullptr, 0, P[LA1_OUT_W], nullptr, TMP2, C, C, Hh, Ww, Hh, Ww, 3, 3, 1, 1, 1);
  bn(TMP2, P[LA1_OUT_G], P[LA1_OUT_BT], nullptr, ATT1, C, HW, 1);

  // ---- local attention 2 (q,k from fre ; v from rgb) ----
  bn(fre, P[LA2_NG], P[LA2_NB], nullptr, A_, C, HW, 0);
  conv(A_, nullptr, 0, P[LA2_QK_W], P[LA2_QK_B], QV, C, 2 * C, Hh, Ww, Hh, Ww, 1, 1, 1, 0, 1);
  bn(rgb, P[LA2_NG], P[LA2_NB], nullptr, A_, C, HW, 0);
  conv(A_, nullptr, 0, P[LA2_V_W], P[LA2_V_B], K1, C, C, Hh, Ww, Hh, Ww, 1, 1, 1, 0, 1);
  multi_attn(QV, 2 * C, 0, QV, 2 * C, C, K1, C, 0, TMP);
  conv(TMP, nullptr, 0, P[LA2_OUT_W], nullptr, TMP2, C, C, Hh, Ww, Hh, Ww, 3, 3, 1, 1, 1);
  bn(TMP2, P[LA2_OUT_G], P[LA2_OUT_BT], nullptr, ATT2, C, HW, 1);

  // ---- fusion: out = rgb + bn(conv3x3(relu(bn(conv1x1(cat))))) ----
  conv(ATT1, ATT2, C, P[CV_W1], nullptr, TMP, 2 * C, C, Hh, Ww, Hh, Ww, 1, 1, 1, 0, 1);
  bn(TMP, P[CV_G1], P[CV_BT1], nullptr, A_, C, HW, 2);
  conv(A_, nullptr, 0, P[CV_W2], nullptr, TMP, C, C, Hh, Ww, Hh, Ww, 3, 3, 1, 1, 1);
  bn(TMP, P[CV_G2], P[CV_BT2], rgb, OUTB, C, HW, 0);

  // ---- FFN 1 ----
  conv(OUTB, nullptr, 0, P[FF1_W1], P[FF1_B1], TMP, C, C, Hh, Ww, Hh, Ww, 3, 3, 1, 2, 2);
  bn(TMP, P[FF1_G1], P[FF1_BT1], nullptr, A_, C, HW, 1);
  conv(A_, nullptr, 0, P[FF1_W2], P[FF1_B2], TMP, C, C, Hh, Ww, Hh, Ww, 3, 3, 1, 1, 1);
  bn(TMP, P[FF1_G2], P[FF1_BT2], nullptr, TMP2, C, HW, 1);
  addv(OUTB, TMP2, OUTB);

  // ---- global attention (SRA, hd=8, sr 7x7 stride 7) ----
  conv(OUTB, nullptr, 0, P[GA_Q_W], P[GA_Q_B], QV, C, C, Hh, Ww, Hh, Ww, 1, 1, 1, 0, 1);
  conv(OUTB, nullptr, 0, P[GA_SR_W], P[GA_SR_B], SR, C, C, Hh, Ww, 8, 8, 7, 7, 7, 0, 1);
  bn(SR, P[GA_NG], P[GA_NB], nullptr, TMP, C, 64, 0);  // TMP (small prefix) = x_
  conv(TMP, nullptr, 0, P[GA_KV_W], P[GA_KV_B], KV, C, 2 * C, 8, 8, 8, 8, 1, 1, 1, 0, 1);
  {
    dim3 grid(HW / 16, 16, Bn);  // 196 query tiles x 16 heads x 16 batch
    global_attn_kernel<<<grid, 32, 0, stream>>>(QV, KV, TMP2, C, HW, 64, 8, SCALE8);
  }
  conv(TMP2, nullptr, 0, P[GA_OUT_W], nullptr, TMP, C, C, Hh, Ww, Hh, Ww, 3, 3, 1, 1, 1);
  bn(TMP, P[GA_OUT_G], P[GA_OUT_BT], OUTB, A_, C, HW, 0);  // A_ = out + g

  // ---- FFN 2 ----
  conv(A_, nullptr, 0, P[FF2_W1], P[FF2_B1], TMP, C, C, Hh, Ww, Hh, Ww, 3, 3, 1, 2, 2);
  bn(TMP, P[FF2_G1], P[FF2_BT1], nullptr, TMP2, C, HW, 1);
  conv(TMP2, nullptr, 0, P[FF2_W2], P[FF2_B2], TMP, C, C, Hh, Ww, Hh, Ww, 3, 3, 1, 1, 1);
  bn(TMP, P[FF2_G2], P[FF2_BT2], nullptr, K1, C, HW, 1);
  addv(A_, K1, (float*)d_out);

  (void)in_sizes; (void)n_in; (void)out_size; (void)ws_size;
}
